// SPINNEncoder_78159814852931
// MI455X (gfx1250) — matrix operand order repaired and checked by reading, not compiled, not run on hardware
//
#include <hip/hip_runtime.h>
#include <hip/hip_bf16.h>
#include <math.h>

typedef __attribute__((ext_vector_type(16))) __bf16 v16bf;
typedef __attribute__((ext_vector_type(8)))  float  v8f;

#define Hc   256
#define Bc   512
#define Sc   256
#define Tc   256
#define Ec   300
#define EPAD 320
#define MT   16          // batch rows per workgroup (WMMA M)
#define NT   (Hc/16)     // 16 n-tiles
#define KT_U (Hc/32)     // 8 k-tiles for the recurrent GEMMs
#define KT_E (EPAD/32)   // 10 k-tiles for the embedding projection
#define RTHREADS 512     // 16 waves: one n-tile per wave in the recurrent kernel

// ---------------------------------------------------------------------------
// A-fragment mapping for V_WMMA_*_16x16x32 (16-bit A, 16x32):
//   lane = m + ((k>>4)&1)<<4 ; pos = k&15    (k = K%32)
// B-fragment mapping (32x16): lane = (Khalf<<4) | N%16, 16 packed K values.
// C/D layout: VGPR r, lane L -> M = r + (L>=16 ? 8 : 0), N = L%16.
// ---------------------------------------------------------------------------

__global__ void pack_wtr_kernel(const float* __restrict__ Wtr,
                                __bf16* __restrict__ dst) {
  int g = blockIdx.x * blockDim.x + threadIdx.x;
  if (g >= KT_E * NT * 32) return;
  int lane = g & 31;
  int nt   = (g >> 5) & 15;
  int kt   = g >> 9;
  int n    = nt * 16 + (lane & 15);
  int kb   = kt * 32 + ((lane >> 4) << 4);
  __bf16* out = dst + (size_t)g * 16;
#pragma unroll
  for (int p = 0; p < 16; ++p) {
    int k = kb + p;
    float v = (k < Ec) ? Wtr[(size_t)k * Hc + n] : 0.f;   // B(k,n) = W_tr[k][n]
    out[p] = (__bf16)v;
  }
}

__global__ void pack_u_kernel(const float* __restrict__ U,
                              __bf16* __restrict__ dst) {
  int g = blockIdx.x * blockDim.x + threadIdx.x;
  if (g >= 8 * KT_U * NT * 32) return;
  int lane = g & 31;
  int nt   = (g >> 5) & 15;
  int kt   = (g >> 9) & 7;
  int u    = g >> 12;
  int n    = nt * 16 + (lane & 15);
  int kb   = kt * 32 + ((lane >> 4) << 4);
  __bf16* out = dst + (size_t)g * 16;
#pragma unroll
  for (int p = 0; p < 16; ++p) {
    int k = kb + p;
    // h @ U.T  =>  B(k,n) = U[u][n][k]
    out[p] = (__bf16)U[((size_t)u * Hc + n) * Hc + k];
  }
}

// ---------------------------------------------------------------------------
// buf[r][n] = emb[seq[r]] @ W_tr + b_tr  (stored bf16), K padded 300->320
// ---------------------------------------------------------------------------
__global__ __launch_bounds__(256) void emb_proj_kernel(
    const float* __restrict__ emb, const int* __restrict__ seq,
    const float* __restrict__ btr, const __bf16* __restrict__ Wpack,
    __bf16* __restrict__ buf) {
  __shared__ __align__(32) __bf16 Afrag[KT_E][32][16];
  __shared__ int tok[MT];
  int tid = threadIdx.x;
  size_t r0 = (size_t)blockIdx.x * MT;
  if (tid < MT) tok[tid] = seq[r0 + tid];
  __syncthreads();
  // stage gathered emb rows as bf16 A-fragments (16 rows x 320 K)
  for (int e = 0; e < (MT * EPAD) / 256; ++e) {
    int idx = e * 256 + tid;
    int m = idx / EPAD;
    int k = idx - m * EPAD;
    float v = (k < Ec) ? emb[(size_t)tok[m] * Ec + k] : 0.f;
    int kt = k >> 5, ko = k & 31;
    Afrag[kt][m + ((ko >> 4) << 4)][ko & 15] = (__bf16)v;
  }
  __syncthreads();
  int lane = tid & 31;
  int wave = tid >> 5;
  for (int nt2 = 0; nt2 < 2; ++nt2) {
    int nt = wave * 2 + nt2;
    v8f acc = (v8f)0.0f;
#pragma unroll
    for (int kt = 0; kt < KT_E; ++kt) {
      v16bf a  = *(const v16bf*)&Afrag[kt][lane][0];
      v16bf bv = *(const v16bf*)(Wpack + (size_t)((kt * NT + nt) * 32 + lane) * 16);
      acc = __builtin_amdgcn_wmma_f32_16x16x32_bf16(false, a, false, bv,
                                                    (short)0, acc, false, false);
    }
    int n = nt * 16 + (lane & 15);
    int mb = (lane >> 4) << 3;
    float bb = btr[n];
#pragma unroll
    for (int r = 0; r < 8; ++r) {
      int m = r + mb;
      buf[(r0 + m) * Hc + n] = (__bf16)(acc[r] + bb);
    }
  }
}

// ---------------------------------------------------------------------------
// The full T=256 shift-reduce TreeLSTM recursion; 16 batch rows per WG,
// 16 waves (512 threads): one 16-col n-tile per wave for max wave residency.
// ---------------------------------------------------------------------------
__global__ __launch_bounds__(RTHREADS) void tree_lstm_kernel(
    const __bf16* __restrict__ buf, const __bf16* __restrict__ Upack,
    const float* __restrict__ bias, const int* __restrict__ trans,
    float* __restrict__ stack, float* __restrict__ out) {
  __shared__ __align__(32) __bf16 Afrag[2][KT_U][32][16];  // [h_l/h_r][kt][lane][16]
  __shared__ float csum[MT][Hc];                           // c_l + c_r
  __shared__ int ptrs_s[MT][Tc + 1];
  __shared__ int depth_s[MT], bptr_s[MT];
  __shared__ int p1_s[MT], p2_s[MT], red_s[MT], bold_s[MT];

  int tid  = threadIdx.x;
  int lane = tid & 31;
  int nt   = tid >> 5;                              // wave id == n-tile id
  int row0 = blockIdx.x * MT;
  const size_t RS = (size_t)(Tc + 1) * (2 * Hc);    // per-row stack stride
  float* stk = stack + (size_t)row0 * RS;

  // init parser state and stack slot 0 (all later reads of "empty" hit slot 0)
  if (tid < MT) { depth_s[tid] = 0; bptr_s[tid] = 0; }
  for (int i = tid; i < MT * (Tc + 1); i += RTHREADS) ((int*)ptrs_s)[i] = 0;
  for (int i = tid; i < MT * 2 * Hc; i += RTHREADS)
    stk[(size_t)(i / (2 * Hc)) * RS + (i % (2 * Hc))] = 0.f;
  __syncthreads();

  for (int t = 1; t <= Tc; ++t) {
    // ---- per-row pointer bookkeeping ------------------------------------
    if (tid < MT) {
      int r = tid;
      int mask = trans[(size_t)(row0 + r) * Tc + (t - 1)];
      int d = depth_s[r];
      p1_s[r] = (d >= 1) ? ptrs_s[r][d - 1] : 0;
      p2_s[r] = (d >= 2) ? ptrs_s[r][d - 2] : 0;
      red_s[r] = mask;
      bold_s[r] = bptr_s[r];
      bptr_s[r] += (1 - mask);
      int dpop = mask ? d - 2 : d;
      if (dpop < 0) dpop = 0;
      ptrs_s[r][dpop] = t;
      depth_s[r] = dpop + 1;
    }
    __syncthreads();
    // ---- stage h_l, h_r as bf16 A-fragments; csum = c_l + c_r -----------
    // 4096 elements over 512 threads; consecutive tids -> consecutive k: coalesced
    for (int idx = tid; idx < MT * Hc; idx += RTHREADS) {
      int m = idx >> 8;
      int k = idx & (Hc - 1);
      const float* rl = stk + (size_t)m * RS + (size_t)p1_s[m] * (2 * Hc);
      const float* rr = stk + (size_t)m * RS + (size_t)p2_s[m] * (2 * Hc);
      float hl = rl[k], hr = rr[k];
      csum[m][k] = rl[Hc + k] + rr[Hc + k];
      int kt = k >> 5, ko = k & 31;
      int ln = m + ((ko >> 4) << 4);
      int ps = ko & 15;
      Afrag[0][kt][ln][ps] = (__bf16)hl;
      Afrag[1][kt][ln][ps] = (__bf16)hr;
    }
    __syncthreads();
    // ---- 4 gates for this wave's n-tile: 2 operands x 8 K-WMMAs each ----
    const int uml[4] = {0, 2, 4, 7};  // U_il, U_fl, U_ol, U_ul
    const int umr[4] = {1, 3, 5, 6};  // U_ir, U_fr, U_or, U_ur
    v8f acc[4];
#pragma unroll
    for (int g = 0; g < 4; ++g) acc[g] = (v8f)0.0f;
#pragma unroll
    for (int side = 0; side < 2; ++side) {
#pragma unroll
      for (int kt = 0; kt < KT_U; ++kt) {
        v16bf a = *(const v16bf*)&Afrag[side][kt][lane][0];
#pragma unroll
        for (int g = 0; g < 4; ++g) {
          int u = side ? umr[g] : uml[g];
          v16bf bv = *(const v16bf*)(Upack +
                      (size_t)(((u * KT_U + kt) * NT + nt) * 32 + lane) * 16);
          acc[g] = __builtin_amdgcn_wmma_f32_16x16x32_bf16(
              false, a, false, bv, (short)0, acc[g], false, false);
        }
      }
    }
    // ---- gating + stack push -------------------------------------------
    int n  = nt * 16 + (lane & 15);
    int mb = (lane >> 4) << 3;
    float bi = bias[n], bf_ = bias[Hc + n], bo = bias[2 * Hc + n], bu = bias[3 * Hc + n];
#pragma unroll
    for (int r = 0; r < 8; ++r) {
      int m = r + mb;
      float vi = 1.f / (1.f + __expf(-(acc[0][r] + bi)));
      float vf = 1.f / (1.f + __expf(-(acc[1][r] + bf_)));
      float vo = 1.f / (1.f + __expf(-(acc[2][r] + bo)));
      float vu = tanhf(acc[3][r] + bu);
      float c = vi * vu + vf * csum[m][n];
      float h = vo * tanhf(c);
      float oh, oc;
      if (red_s[m]) { oh = h; oc = c; }
      else {
        float bv = (float)buf[((size_t)(row0 + m) * Sc + bold_s[m]) * Hc + n];
        oh = bv; oc = bv;   // buffer_top duplicated across [h|c] halves
      }
      float* dst = stk + (size_t)m * RS + (size_t)t * (2 * Hc);
      dst[n] = oh;
      dst[Hc + n] = oc;
      if (t == Tc) {
        out[(size_t)(row0 + m) * (2 * Hc) + n] = oh;
        out[(size_t)(row0 + m) * (2 * Hc) + Hc + n] = oc;
      }
    }
    __threadfence_block();   // stack stores (WGP scope) visible to next step
    __syncthreads();
  }
}

// ---------------------------------------------------------------------------
extern "C" void kernel_launch(void* const* d_in, const int* in_sizes, int n_in,
                              void* d_out, int out_size, void* d_ws, size_t ws_size,
                              hipStream_t stream) {
  const float* emb  = (const float*)d_in[0];
  const float* Wtr  = (const float*)d_in[1];
  const float* btr  = (const float*)d_in[2];
  // d_in[3] = Wx : unused (tracking input x == 0 -> Wx terms vanish)
  const float* U    = (const float*)d_in[4];
  const float* bias = (const float*)d_in[5];
  const int*   seq  = (const int*)d_in[6];
  const int*   trn  = (const int*)d_in[7];
  float* out = (float*)d_out;

  char* ws = (char*)d_ws;
  size_t offW = 0;
  size_t szW  = (size_t)KT_E * NT * 32 * 16 * 2;        // 160 KB packed W_tr
  size_t offU = offW + szW;
  size_t szU  = (size_t)8 * KT_U * NT * 32 * 16 * 2;    // 1 MB packed U
  size_t offB = (offU + szU + 255) & ~(size_t)255;
  size_t szB  = (size_t)Bc * Sc * Hc * 2;               // 64 MB bf16 buf
  size_t offS = (offB + szB + 255) & ~(size_t)255;
  size_t szS  = (size_t)Bc * (Tc + 1) * (2 * Hc) * 4;   // 257 MB f32 stack
  if (offS + szS > ws_size) return;                     // workspace too small

  __bf16* Wpack = (__bf16*)(ws + offW);
  __bf16* Upack = (__bf16*)(ws + offU);
  __bf16* buf   = (__bf16*)(ws + offB);
  float*  stk   = (float*)(ws + offS);

  pack_wtr_kernel<<<(KT_E * NT * 32 + 255) / 256, 256, 0, stream>>>(Wtr, Wpack);
  pack_u_kernel<<<(8 * KT_U * NT * 32 + 255) / 256, 256, 0, stream>>>(U, Upack);
  emb_proj_kernel<<<(Bc * Sc) / MT, 256, 0, stream>>>(emb, seq, btr, Wpack, buf);
  tree_lstm_kernel<<<Bc / MT, RTHREADS, 0, stream>>>(buf, Upack, bias, trn, stk, out);
}
